// SparseTriXFFN_17506286698974
// MI455X (gfx1250) — matrix operand order repaired and checked by reading, not compiled, not run on hardware
//
#include <hip/hip_runtime.h>
#include <hip/hip_bf16.h>
#include <stddef.h>

typedef __attribute__((ext_vector_type(16))) __bf16 v16bf;
typedef __attribute__((ext_vector_type(8)))  __bf16 v8bf;
typedef __attribute__((ext_vector_type(8)))  float  v8f;

#define D_MODEL 1024
#define D_FF    4096
#define N_TOK   8192
#define KT      4
#define TS      1024
#define HSTRIDE 1032   // padded LDS row stride (bf16 elems) to spread banks

// ws layout (bytes)
#define OFF_UWB    ((size_t)0)            // 8 MiB  sign(up_w)  bf16 [k][t][d]
#define OFF_DWB    ((size_t)8  << 20)     // 8 MiB  sign(down_w) bf16 [d][4*ts]
#define OFF_XBF    ((size_t)16 << 20)     // (8192+1)x1024 bf16; row 8192 = zeros
#define OFF_SIGS   ((size_t)33 << 20)     // 16 KiB
#define OFF_CNT    (OFF_SIGS + 16384)     // 256 B
#define OFF_BUCKET (OFF_CNT + 256)        // 128 KiB

// ---------------- kernel 1: binarize weights to bf16 (+1/-1/0) ----------------
__global__ void trix_sign_kernel(const float* __restrict__ up_w,
                                 const float* __restrict__ down_w,
                                 __bf16* __restrict__ uwB,
                                 __bf16* __restrict__ dwB) {
    const int n = D_FF * D_MODEL;  // 4M elements each
    for (int i = blockIdx.x * blockDim.x + threadIdx.x; i < n;
         i += gridDim.x * blockDim.x) {
        float u = up_w[i];
        float d = down_w[i];
        uwB[i] = (__bf16)((u > 0.f) ? 1.f : ((u < 0.f) ? -1.f : 0.f));
        dwB[i] = (__bf16)((d > 0.f) ? 1.f : ((d < 0.f) ? -1.f : 0.f));
    }
}

// ---------------- kernel 1b: x -> bf16 (plus zeroed sentinel row 8192) --------
__global__ void trix_xbf_kernel(const float* __restrict__ x,
                                __bf16* __restrict__ xbf) {
    const int n = (N_TOK + 1) * D_MODEL;
    for (int i = blockIdx.x * blockDim.x + threadIdx.x; i < n;
         i += gridDim.x * blockDim.x) {
        float v = (i < N_TOK * D_MODEL) ? x[i] : 0.f;
        xbf[i] = (__bf16)v;
    }
}

// ---------------- kernel 2: tile signatures sigs[k][d] = sign(mean_t up_w) ----
__global__ void trix_sigs_kernel(const float* __restrict__ up_w,
                                 float* __restrict__ sigs) {
    int tid = blockIdx.x * blockDim.x + threadIdx.x;   // 4096 threads
    if (tid >= KT * D_MODEL) return;
    int k = tid / D_MODEL;
    int d = tid % D_MODEL;
    const float* base = up_w + (size_t)k * TS * D_MODEL + d;
    float s = 0.f;
    for (int t = 0; t < TS; ++t) s += base[(size_t)t * D_MODEL];
    sigs[tid] = (s > 0.f) ? 1.f : ((s < 0.f) ? -1.f : 0.f);
}

// ---------------- kernel 3: zero bucket counters ------------------------------
__global__ void trix_zero_counts(int* __restrict__ counts) {
    if (threadIdx.x < KT) counts[threadIdx.x] = 0;
}

// ---------------- kernel 4: router (one wave32 per token) ---------------------
__global__ void trix_router_kernel(const float* __restrict__ x,
                                   const float* __restrict__ sigs,
                                   float* __restrict__ gate_out,
                                   int* __restrict__ counts,
                                   int* __restrict__ bucket) {
    int wave = threadIdx.x >> 5;
    int lane = threadIdx.x & 31;
    int tok  = blockIdx.x * 8 + wave;          // 256 thr = 8 waves = 8 tokens
    if (tok >= N_TOK) return;
    const float* xr = x + (size_t)tok * D_MODEL;
    float s0 = 0.f, s1 = 0.f, s2 = 0.f, s3 = 0.f;
    for (int d = lane; d < D_MODEL; d += 32) {
        float xv = xr[d];
        s0 += xv * sigs[0 * D_MODEL + d];
        s1 += xv * sigs[1 * D_MODEL + d];
        s2 += xv * sigs[2 * D_MODEL + d];
        s3 += xv * sigs[3 * D_MODEL + d];
    }
    #pragma unroll
    for (int off = 16; off >= 1; off >>= 1) {
        s0 += __shfl_xor(s0, off, 32);
        s1 += __shfl_xor(s1, off, 32);
        s2 += __shfl_xor(s2, off, 32);
        s3 += __shfl_xor(s3, off, 32);
    }
    if (lane == 0) {
        float sc[4] = {s0, s1, s2, s3};
        int best = 0;
        #pragma unroll
        for (int j = 1; j < KT; ++j)
            if (sc[j] > sc[best]) best = j;     // strict > : first max wins
        #pragma unroll
        for (int j = 0; j < KT; ++j)
            gate_out[(size_t)tok * KT + j] = (j == best) ? 1.f : 0.f;
        int pos = atomicAdd(&counts[best], 1);
        bucket[best * N_TOK + pos] = tok;
    }
}

// -------- device helpers for the WMMA pipeline --------------------------------
__device__ __forceinline__ v16bf trix_load_a(const __bf16* row, int kk, int half) {
    v8bf lo = *(const v8bf*)(row + kk + half * 16);
    v8bf hi = *(const v8bf*)(row + kk + half * 16 + 8);
    v16bf a;
    #pragma unroll
    for (int i = 0; i < 8; ++i) { a[i] = lo[i]; a[i + 8] = hi[i]; }
    return a;
}

__device__ __forceinline__ void trix_load_bp(v16bf (&b)[8], const __bf16* brow,
                                             size_t stride, int kk) {
    #pragma unroll
    for (int nt = 0; nt < 8; ++nt)
        b[nt] = *(const v16bf*)(brow + (size_t)nt * 16 * stride + kk);
}

__device__ __forceinline__ void trix_pf_bp(const __bf16* brow, size_t stride,
                                           int kk) {
    #pragma unroll
    for (int nt = 0; nt < 8; ++nt)
        __builtin_prefetch(brow + (size_t)nt * 16 * stride + kk, 0, 1);
}

__device__ __forceinline__ void trix_wmma8(v8f (&acc)[8], v16bf a,
                                           const v16bf (&b)[8]) {
    #pragma unroll
    for (int nt = 0; nt < 8; ++nt)
        acc[nt] = __builtin_amdgcn_wmma_f32_16x16x32_bf16(
            false, a, false, b[nt], (short)0, acc[nt], false, false);
}

// Double-buffered 16xK GEMM panel: acc[nt] += A(16xK) * B(Kx16)_nt, K = kdim.
__device__ __forceinline__ void trix_gemm_panel(v8f (&acc)[8],
                                                const __bf16* arow,
                                                const __bf16* brow,
                                                size_t stride, int kdim,
                                                int half) {
    v16bf b0[8], b1[8];
    trix_load_bp(b0, brow, stride, 0);
    v16bf a0 = trix_load_a(arow, 0, half);
    for (int kk = 0; kk < kdim; kk += 64) {
        // stage next panel's loads before consuming the current one
        trix_load_bp(b1, brow, stride, kk + 32);
        v16bf a1 = trix_load_a(arow, kk + 32, half);
        trix_pf_bp(brow, stride, kk + 96 < kdim ? kk + 96 : kk + 32);
        trix_wmma8(acc, a0, b0);                 // hides b1/a1 load latency
        if (kk + 64 < kdim) {
            trix_load_bp(b0, brow, stride, kk + 64);
            a0 = trix_load_a(arow, kk + 64, half);
        }
        trix_wmma8(acc, a1, b1);                 // hides b0/a0 load latency
    }
}

// ---------------- kernel 5: fused winner-gated FFN GEMM (bf16 WMMA) -----------
__global__ void __launch_bounds__(256)
trix_ffn_kernel(const __bf16* __restrict__ xbf,   // [8193][1024], row 8192 = 0
                const float* __restrict__ up_scales,
                const float* __restrict__ down_scales,
                const __bf16* __restrict__ uwB,   // [k][t][d]  sign(up_w)
                const __bf16* __restrict__ dwB,   // [d][4*ts]  sign(down_w)
                const int* __restrict__ counts,
                const int* __restrict__ bucket,
                float* __restrict__ out) {
    __shared__ __align__(32) __bf16 Hs[16 * HSTRIDE];   // hidden tile, 33KB
    __shared__ int toks[16];

    const int k     = blockIdx.y;
    const int chunk = blockIdx.x;
    const int cnt   = counts[k];
    const int c0    = chunk * 16;
    if (c0 >= cnt) return;

    if (threadIdx.x < 16)
        toks[threadIdx.x] = (c0 + threadIdx.x < cnt)
                                ? bucket[k * N_TOK + c0 + threadIdx.x]
                                : N_TOK;                 // sentinel -> zero row
    __syncthreads();

    const int wave = threadIdx.x >> 5;
    const int lane = threadIdx.x & 31;
    const int n    = lane & 15;       // A row (M) / B,D column (N) for this lane
    const int half = lane >> 4;

    // ---- up projection: H[16, 1024] = relu((X @ uwB_k^T) * us) -> LDS bf16 ---
    {
        const __bf16* xrow = xbf + (size_t)toks[n] * D_MODEL;
        const __bf16* brow = uwB + (size_t)k * TS * D_MODEL
                             + (size_t)(wave * 128 + n) * D_MODEL + half * 16;
        const float*  us   = up_scales + k * TS;

        v8f acc[8];
        #pragma unroll
        for (int i = 0; i < 8; ++i) acc[i] = 0.f;

        trix_gemm_panel(acc, xrow, brow, D_MODEL, D_MODEL, half);

        // epilogue: scale, relu, pack to LDS bf16
        #pragma unroll
        for (int nt = 0; nt < 8; ++nt) {
            int t0 = (wave * 8 + nt) * 16;
            float sc = us[t0 + n];
            #pragma unroll
            for (int r = 0; r < 8; ++r) {
                int M = r + half * 8;
                float v = acc[nt][r] * sc;
                v = v > 0.f ? v : 0.f;
                Hs[M * HSTRIDE + t0 + n] = (__bf16)v;
            }
        }
    }
    __syncthreads();

    // ---- down projection: out[16, 1024] = (H @ dwB_k^T) * ds -----------------
    {
        // element (d,t) of tile k lives at dwB[d*D_FF + k*TS + t]
        const __bf16* brow = dwB + (size_t)(wave * 128 + n) * D_FF
                             + k * TS + half * 16;
        const __bf16* hrow = Hs + n * HSTRIDE;

        v8f acc[8];
        #pragma unroll
        for (int i = 0; i < 8; ++i) acc[i] = 0.f;

        trix_gemm_panel(acc, hrow, brow, D_FF, TS, half);

        // epilogue: scale by down_scales, scatter to winner token rows
        #pragma unroll
        for (int nt = 0; nt < 8; ++nt) {
            int d0 = (wave * 8 + nt) * 16;
            float dsc = down_scales[d0 + n];
            #pragma unroll
            for (int r = 0; r < 8; ++r) {
                int M = r + half * 8;
                if (c0 + M < cnt)
                    out[(size_t)toks[M] * D_MODEL + d0 + n] = acc[nt][r] * dsc;
            }
        }
    }
}

// ---------------- host launcher ----------------------------------------------
extern "C" void kernel_launch(void* const* d_in, const int* in_sizes, int n_in,
                              void* d_out, int out_size, void* d_ws, size_t ws_size,
                              hipStream_t stream) {
    (void)in_sizes; (void)n_in; (void)out_size; (void)ws_size;
    const float* x           = (const float*)d_in[0];
    const float* up_w        = (const float*)d_in[1];
    const float* up_scales   = (const float*)d_in[2];
    const float* down_w      = (const float*)d_in[3];
    const float* down_scales = (const float*)d_in[4];

    float* out  = (float*)d_out;
    float* gate = out + (size_t)N_TOK * D_MODEL;

    char* ws = (char*)d_ws;
    __bf16* uwB    = (__bf16*)(ws + OFF_UWB);
    __bf16* dwB    = (__bf16*)(ws + OFF_DWB);
    __bf16* xbf    = (__bf16*)(ws + OFF_XBF);
    float*  sigs   = (float*) (ws + OFF_SIGS);
    int*    counts = (int*)   (ws + OFF_CNT);
    int*    bucket = (int*)   (ws + OFF_BUCKET);

    trix_sign_kernel  <<<4096, 256, 0, stream>>>(up_w, down_w, uwB, dwB);
    trix_xbf_kernel   <<<8192, 256, 0, stream>>>(x, xbf);
    trix_sigs_kernel  <<<16,   256, 0, stream>>>(up_w, sigs);
    trix_zero_counts  <<<1,     64, 0, stream>>>(counts);
    trix_router_kernel<<<N_TOK / 8, 256, 0, stream>>>(x, sigs, gate, counts, bucket);

    dim3 grid(N_TOK / 16, KT);   // 512 chunks x 4 tiles; inactive chunks exit
    trix_ffn_kernel<<<grid, 256, 0, stream>>>(xbf, up_scales, down_scales,
                                              uwB, dwB, counts, bucket, out);
}
